// VanillaGAT_34626026340832
// MI455X (gfx1250) — matrix operand order repaired and checked by reading, not compiled, not run on hardware
//
#include <hip/hip_runtime.h>

typedef __attribute__((ext_vector_type(16))) __bf16 v16bf;
typedef __attribute__((ext_vector_type(8)))  __bf16 v8bf;
typedef __attribute__((ext_vector_type(8)))  float  v8f;

#define SHUF16(x, y) __builtin_shufflevector((x), (y), 0, 1, 2, 3, 4, 5, 6, 7, \
                                             8, 9, 10, 11, 12, 13, 14, 15)

// ---------------------------------------------------------------------------
// Signed float atomic max: int-max for non-negative, uint-min for negative.
// Works with -inf (0xFF800000) initialization.
// ---------------------------------------------------------------------------
__device__ __forceinline__ void atomicMaxFloat(float* addr, float val) {
  if (val >= 0.0f) {
    atomicMax((int*)addr, __float_as_int(val));
  } else {
    atomicMin((unsigned int*)addr, __float_as_uint(val));
  }
}

__device__ __forceinline__ void edge_nodes(const long long* __restrict__ src,
                                           const long long* __restrict__ dst,
                                           int e, int E, int& s, int& d) {
  if (e < E) { s = (int)src[e]; d = (int)dst[e]; }
  else       { s = e - E; d = e - E; }   // appended self-loops
}

// ---------------------------------------------------------------------------
// Operand packing for WMMA: f32 -> bf16 hi + bf16 lo (error compensation),
// K zero-padded to Kp (multiple of 32) so the GEMM loop has no guards.
// ---------------------------------------------------------------------------
__global__ void k_pack_rows(const float* __restrict__ A, __bf16* __restrict__ hi,
                            __bf16* __restrict__ lo, int N, int Ka, int Kp) {
  int idx = blockIdx.x * blockDim.x + threadIdx.x;
  if (idx >= N * Kp) return;
  int n = idx / Kp, k = idx - n * Kp;
  float x = (k < Ka) ? A[(size_t)n * Ka + k] : 0.0f;
  __bf16 h = (__bf16)x;
  hi[idx] = h;
  lo[idx] = (__bf16)(x - (float)h);
}

// W[Ka,M] -> transposed Wt[M,Kp] so each lane's B fragment is contiguous.
__global__ void k_pack_wT(const float* __restrict__ W, __bf16* __restrict__ hi,
                          __bf16* __restrict__ lo, int Ka, int M, int Kp) {
  int idx = blockIdx.x * blockDim.x + threadIdx.x;
  if (idx >= M * Kp) return;
  int m = idx / Kp, k = idx - m * Kp;
  float x = (k < Ka) ? W[(size_t)k * M + m] : 0.0f;
  __bf16 h = (__bf16)x;
  hi[idx] = h;
  lo[idx] = (__bf16)(x - (float)h);
}

// ---------------------------------------------------------------------------
// C[N,M] = act(A @ W + bias) from packed bf16 hi/lo operands.
// One 16x16 tile per wave32; inner loop: 8x global_load_b128 + 3 WMMAs.
// Requires N % 16 == 0, Kp % 32 == 0, M % 16 == 0. act: 0=none, 1=relu.
// ---------------------------------------------------------------------------
__global__ void k_gemm_wmma(const __bf16* __restrict__ Ahi, const __bf16* __restrict__ Alo,
                            const __bf16* __restrict__ Bhi, const __bf16* __restrict__ Blo,
                            const float* __restrict__ bias, float* __restrict__ C,
                            int N, int Kp, int M, int act) {
  const int wave = blockIdx.x * (blockDim.x >> 5) + (threadIdx.x >> 5);
  const int colTiles = M >> 4;
  if (wave >= (N >> 4) * colTiles) return;   // wave-uniform exit, EXEC stays full
  const int rowTile = wave / colTiles;
  const int colTile = wave - rowTile * colTiles;
  const int lane = threadIdx.x & 31;
  const int half = lane >> 4;
  const int r    = lane & 15;
  // A 16x32 bf16 layout: lane row = r; lanes 0-15 K {0-7,16-23}, lanes 16-31 +8
  const size_t aBase = (size_t)((rowTile << 4) + r) * Kp + (half << 3);
  // B 32x16 bf16 layout: lane col = r; lanes 0-15 K {0-7,16-23}, lanes 16-31 +8
  const size_t bBase = (size_t)((colTile << 4) + r) * Kp + (half << 3);

  v8f acc = {};
  for (int k0 = 0; k0 < Kp; k0 += 32) {
    const __bf16* ah = Ahi + aBase + k0;
    const __bf16* al = Alo + aBase + k0;
    const __bf16* bh = Bhi + bBase + k0;
    const __bf16* bl = Blo + bBase + k0;
    v8bf ah0 = *(const v8bf*)(ah);       // K = k0 + half*8 + 0..7
    v8bf ah1 = *(const v8bf*)(ah + 16);  // K = k0 + 16 + half*8 + 0..7
    v8bf al0 = *(const v8bf*)(al);
    v8bf al1 = *(const v8bf*)(al + 16);
    v8bf bh0 = *(const v8bf*)(bh);       // K = k0 + half*8 + 0..7
    v8bf bh1 = *(const v8bf*)(bh + 16);  // K = k0 + 16 + half*8 + 0..7
    v8bf bl0 = *(const v8bf*)(bl);
    v8bf bl1 = *(const v8bf*)(bl + 16);
    v16bf aH = SHUF16(ah0, ah1);
    v16bf aL = SHUF16(al0, al1);
    v16bf bH = SHUF16(bh0, bh1);
    v16bf bL = SHUF16(bl0, bl1);
    acc = __builtin_amdgcn_wmma_f32_16x16x32_bf16(false, aH, false, bH,
                                                  (short)0, acc, false, false);
    acc = __builtin_amdgcn_wmma_f32_16x16x32_bf16(false, aL, false, bH,
                                                  (short)0, acc, false, false);
    acc = __builtin_amdgcn_wmma_f32_16x16x32_bf16(false, aH, false, bL,
                                                  (short)0, acc, false, false);
  }

  const int nOut = (colTile << 4) + r;          // C/D layout: N = lane&15
  const float bv = bias ? bias[nOut] : 0.0f;
#pragma unroll
  for (int v = 0; v < 8; ++v) {
    int m = (rowTile << 4) + v + (half << 3);   // C/D layout: M = vgpr + half*8
    float val = acc[v] + bv;
    if (act == 1) val = fmaxf(val, 0.0f);
    C[(size_t)m * M + nOut] = val;
  }
}

// ---------------------------------------------------------------------------
// alpha_s / alpha_d projections: [N,H] dots of h[n,hd,:] with a_src/a_dst[hd,:]
// ---------------------------------------------------------------------------
__global__ void k_alpha(const float* __restrict__ h,
                        const float* __restrict__ asrc,
                        const float* __restrict__ adst,
                        float* __restrict__ als, float* __restrict__ ald, int N) {
  int idx = blockIdx.x * blockDim.x + threadIdx.x;
  if (idx >= N * 8) return;
  int n = idx >> 3, hd = idx & 7;
  const float* hp = h + (size_t)n * 96 + hd * 12;
  const float* sa = asrc + hd * 12;
  const float* da = adst + hd * 12;
  float ss = 0.0f, dd = 0.0f;
#pragma unroll
  for (int c = 0; c < 12; ++c) { float hv = hp[c]; ss += hv * sa[c]; dd += hv * da[c]; }
  als[idx] = ss;
  ald[idx] = dd;
}

__global__ void k_fill(float* __restrict__ p, float v, int n) {
  int i = blockIdx.x * blockDim.x + threadIdx.x;
  if (i < n) p[i] = v;
}

// Pass 1: e = leaky_relu(alpha_s[src]+alpha_d[dst]); segment max over dst.
__global__ void k_edge_max(const long long* __restrict__ src,
                           const long long* __restrict__ dst,
                           const float* __restrict__ als,
                           const float* __restrict__ ald,
                           float* __restrict__ ee, float* __restrict__ emax,
                           int E, int ET) {
  int idx = blockIdx.x * blockDim.x + threadIdx.x;
  if (idx >= ET * 8) return;
  int e = idx >> 3, hd = idx & 7;
  int s, d; edge_nodes(src, dst, e, E, s, d);
  float ev = als[s * 8 + hd] + ald[d * 8 + hd];
  ev = (ev > 0.0f) ? ev : 0.2f * ev;          // leaky_relu, slope 0.2
  ee[idx] = ev;
  atomicMaxFloat(emax + d * 8 + hd, ev);
}

// Pass 2: ee = exp(e - emax[dst]); segment sum into denom.
__global__ void k_edge_exp(const long long* __restrict__ src,
                           const long long* __restrict__ dst,
                           float* __restrict__ ee,
                           const float* __restrict__ emax,
                           float* __restrict__ den, int E, int ET) {
  int idx = blockIdx.x * blockDim.x + threadIdx.x;
  if (idx >= ET * 8) return;
  int e = idx >> 3, hd = idx & 7;
  int s, d; edge_nodes(src, dst, e, E, s, d);
  float x = expf(ee[idx] - emax[d * 8 + hd]);
  ee[idx] = x;
  atomicAdd(den + d * 8 + hd, x);
}

// Pass 3: normalize to softmax weights.
__global__ void k_edge_norm(const long long* __restrict__ src,
                            const long long* __restrict__ dst,
                            float* __restrict__ ee,
                            const float* __restrict__ den, int E, int ET) {
  int idx = blockIdx.x * blockDim.x + threadIdx.x;
  if (idx >= ET * 8) return;
  int e = idx >> 3, hd = idx & 7;
  int s, d; edge_nodes(src, dst, e, E, s, d);
  ee[idx] = ee[idx] / den[d * 8 + hd];
}

// Pass 4: out[dst,:] += h[src,:] * alpha.  4 channels per thread (4 | 12, so a
// float4 group never straddles a head): b128 gather + 4 f32 atomic scatters.
__global__ void k_edge_aggr(const long long* __restrict__ src,
                            const long long* __restrict__ dst,
                            const float* __restrict__ h,
                            const float* __restrict__ ee,
                            float* __restrict__ acc, int E, int ET) {
  int idx = blockIdx.x * blockDim.x + threadIdx.x;
  if (idx >= ET * 24) return;
  int e  = idx / 24;
  int cq = idx - e * 24;
  int c  = cq << 2;
  int s, d; edge_nodes(src, dst, e, E, s, d);
  float a = ee[e * 8 + cq / 3];
  float4 hv = *(const float4*)(h + (size_t)s * 96 + c);
  float* ap = acc + (size_t)d * 96 + c;
  atomicAdd(ap + 0, hv.x * a);
  atomicAdd(ap + 1, hv.y * a);
  atomicAdd(ap + 2, hv.z * a);
  atomicAdd(ap + 3, hv.w * a);
}

// ELU + bias, in place (layers 0 and 1, concat mode).
__global__ void k_elu_bias(float* __restrict__ acc, const float* __restrict__ bias, int N) {
  int idx = blockIdx.x * blockDim.x + threadIdx.x;
  if (idx >= N * 96) return;
  int c = idx % 96;
  float v = acc[idx] + bias[c];
  acc[idx] = (v > 0.0f) ? v : (expf(v) - 1.0f);
}

// Head mean + bias (layer 2, concat=False).
__global__ void k_mean_bias(const float* __restrict__ acc, const float* __restrict__ bias,
                            float* __restrict__ mid, int N) {
  int idx = blockIdx.x * blockDim.x + threadIdx.x;
  if (idx >= N * 12) return;
  int n = idx / 12;
  int c = idx - n * 12;
  float sum = 0.0f;
#pragma unroll
  for (int hd = 0; hd < 8; ++hd) sum += acc[(size_t)n * 96 + hd * 12 + c];
  mid[idx] = sum * 0.125f + bias[c];
}

// ---------------------------------------------------------------------------
extern "C" void kernel_launch(void* const* d_in, const int* in_sizes, int n_in,
                              void* d_out, int out_size, void* d_ws, size_t ws_size,
                              hipStream_t stream) {
  const float*     nf  = (const float*)d_in[0];
  const long long* ei  = (const long long*)d_in[1];
  const float*     wIn = (const float*)d_in[2];
  const float*     bIn = (const float*)d_in[3];
  const float* Wl[3]  = {(const float*)d_in[4],  (const float*)d_in[8],  (const float*)d_in[12]};
  const float* Asr[3] = {(const float*)d_in[5],  (const float*)d_in[9],  (const float*)d_in[13]};
  const float* Ads[3] = {(const float*)d_in[6],  (const float*)d_in[10], (const float*)d_in[14]};
  const float* Bi[3]  = {(const float*)d_in[7],  (const float*)d_in[11], (const float*)d_in[15]};
  const float* wc1 = (const float*)d_in[16];
  const float* bc1 = (const float*)d_in[17];
  const float* wc2 = (const float*)d_in[18];
  const float* bc2 = (const float*)d_in[19];

  const int N  = in_sizes[0] / 128;   // 50000 (multiple of 16)
  const int E  = in_sizes[1] / 2;     // 800000
  const int ET = E + N;               // + self loops
  const int DH = 96, H = 8, CC = 12;

  // Workspace carve-out (~129 MB)
  float* ws = (float*)d_ws;
  size_t off = 0;
  float* xbuf = ws + off; off += (size_t)N * DH;
  float* hbuf = ws + off; off += (size_t)N * DH;
  float* accb = ws + off; off += (size_t)N * DH;
  float* als  = ws + off; off += (size_t)N * H;
  float* ald  = ws + off; off += (size_t)N * H;
  float* emax = ws + off; off += (size_t)N * H;
  float* den  = ws + off; off += (size_t)N * H;
  float* ee   = ws + off; off += (size_t)ET * H;
  float* mid  = ws + off; off += (size_t)N * CC;
  float* c1   = ws + off; off += (size_t)N * 48;
  __bf16* apkh = (__bf16*)(ws + off); off += (size_t)N * 64;   // N*128 bf16
  __bf16* apkl = (__bf16*)(ws + off); off += (size_t)N * 64;
  __bf16* wpkh = (__bf16*)(ws + off); off += 8192;             // 128*96 bf16 + pad
  __bf16* wpkl = (__bf16*)(ws + off); off += 8192;
  (void)ws_size; (void)n_in; (void)out_size;

  const long long* srcI = ei;
  const long long* dstI = ei + E;

  const int TB = 256;                 // 8 waves / block
  auto blocks = [&](int n) { return (n + TB - 1) / TB; };
  auto gemm = [&](const float* A, int Ka, const float* W, int M,
                  const float* bias, float* C, int act) {
    int Kp = (Ka + 31) & ~31;
    k_pack_rows<<<blocks(N * Kp), TB, 0, stream>>>(A, apkh, apkl, N, Ka, Kp);
    k_pack_wT  <<<blocks(M * Kp), TB, 0, stream>>>(W, wpkh, wpkl, Ka, M, Kp);
    int tiles = (N / 16) * (M / 16);
    k_gemm_wmma<<<(tiles + 7) / 8, TB, 0, stream>>>(apkh, apkl, wpkh, wpkl,
                                                    bias, C, N, Kp, M, act);
  };

  // Input MLP: x = relu(nf @ w_in + b_in)
  gemm(nf, 128, wIn, DH, bIn, xbuf, 1);

  float* cur = xbuf;
  float* nxt = accb;
  const int nh = N * H, nacc = N * DH, eh = ET * H, ec4 = ET * 24;
  for (int l = 0; l < 3; ++l) {
    gemm(cur, DH, Wl[l], DH, nullptr, hbuf, 0);    // h = x @ W
    k_alpha<<<blocks(nh), TB, 0, stream>>>(hbuf, Asr[l], Ads[l], als, ald, N);
    k_fill<<<blocks(nh),   TB, 0, stream>>>(emax, -__builtin_inff(), nh);
    k_fill<<<blocks(nh),   TB, 0, stream>>>(den, 0.0f, nh);
    k_fill<<<blocks(nacc), TB, 0, stream>>>(nxt, 0.0f, nacc);
    k_edge_max <<<blocks(eh),  TB, 0, stream>>>(srcI, dstI, als, ald, ee, emax, E, ET);
    k_edge_exp <<<blocks(eh),  TB, 0, stream>>>(srcI, dstI, ee, emax, den, E, ET);
    k_edge_norm<<<blocks(eh),  TB, 0, stream>>>(srcI, dstI, ee, den, E, ET);
    k_edge_aggr<<<blocks(ec4), TB, 0, stream>>>(srcI, dstI, hbuf, ee, nxt, E, ET);
    if (l < 2) {
      k_elu_bias<<<blocks(nacc), TB, 0, stream>>>(nxt, Bi[l], N);
      float* t = cur; cur = nxt; nxt = t;          // ping-pong
    } else {
      k_mean_bias<<<blocks(N * CC), TB, 0, stream>>>(nxt, Bi[l], mid, N);
    }
  }

  // Head MLP: relu(mid @ wc1 + bc1) @ wc2 + bc2
  gemm(mid, 12, wc1, 48, bc1, c1, 1);
  gemm(c1, 48, wc2, 16, bc2, (float*)d_out, 0);
}
